// MPNN_sparse_63780264346299
// MI455X (gfx1250) — matrix operand order repaired and checked by reading, not compiled, not run on hardware
//
#include <hip/hip_runtime.h>

// ---------------------------------------------------------------------------
// MPNN sparse step for gfx1250 (MI455X)
//  - bf16 WMMA GEMMs, f32 accumulation
//  - persistent message kernel: weights resident in LDS (320 KB/WGP),
//    double-buffered edge gather via GLOBAL_LOAD_ASYNC_TO_LDS_B128 (ASYNCcnt)
//  - f32 atomic scatter-add for segment_sum
// ---------------------------------------------------------------------------

typedef __attribute__((ext_vector_type(16))) __bf16 v16bf;
typedef __attribute__((ext_vector_type(8)))  float  v8f;

#define D_IN   128
#define D_DEG  16
#define D_FEAT 144          // D_IN + D_DEG
#define K_A    288          // msg MLP K (=2*D_FEAT); upd MLP K (272) zero-padded to 288
#define D_HID  256
#define D_MSG  128
#define D_UP   128
#define MT     32           // rows (edges/nodes) per tile

union Frag {
    v16bf v;
    uint4 q[2];
};

__device__ __forceinline__ unsigned short f2bf(float f) {
    unsigned u = __float_as_uint(f);
    u += 0x7FFFu + ((u >> 16) & 1u);          // round-to-nearest-even
    return (unsigned short)(u >> 16);
}

__device__ __forceinline__ v8f wmma_bf16(const Frag& a, const Frag& b, v8f c) {
    return __builtin_amdgcn_wmma_f32_16x16x32_bf16(
        /*neg_a=*/false, a.v, /*neg_b=*/false, b.v,
        /*c_mod=*/(short)0, c, /*reuse_a=*/false, /*reuse_b=*/false);
}

// A fragment: 16x32 bf16 tile, row-major, row stride ld elems.
// lane L -> row L&15; lanes 0-15: K {0..7,16..23}; lanes 16-31: K {8..15,24..31}.
__device__ __forceinline__ Frag load_a(const unsigned short* base, int ld, int lane) {
    Frag f;
    const unsigned short* p = base + (lane & 15) * ld + ((lane >> 4) ? 8 : 0);
    f.q[0] = *(const uint4*)(p);
    f.q[1] = *(const uint4*)(p + 16);
    return f;
}

// B fragment: 32x16 bf16 tile from WT stored [out_col][K] (row stride ld elems).
// lane L -> col L&15; lanes 0-15: K {0..7,16..23}; lanes 16-31: K {8..15,24..31}.
__device__ __forceinline__ Frag load_b(const unsigned short* base, int ld, int lane) {
    Frag f;
    const unsigned short* p = base + (lane & 15) * ld + ((lane >> 4) ? 8 : 0);
    f.q[0] = *(const uint4*)(p);
    f.q[1] = *(const uint4*)(p + 16);
    return f;
}

// Async gather: 16 B per lane, global -> LDS, tracked by ASYNCcnt.
__device__ __forceinline__ void async_g2l_b128(void* lds_dst, const void* gsrc) {
    unsigned off = (unsigned)(unsigned long long)lds_dst;   // low 32 bits == LDS offset
    asm volatile("global_load_async_to_lds_b128 %0, %1, off"
                 :: "v"(off), "v"(gsrc)
                 : "memory");
}

__device__ __forceinline__ void wait_asynccnt_le(int n) {
    if (n == 0)      asm volatile("s_wait_asynccnt 0x0" ::: "memory");
    else if (n == 2) asm volatile("s_wait_asynccnt 0x2" ::: "memory");
    else             asm volatile("s_wait_asynccnt 0x3" ::: "memory");
}

// ------------------------------- prep kernels ------------------------------

__global__ void build_h_kernel(const float* __restrict__ x,
                               const float* __restrict__ deg,
                               unsigned short* __restrict__ hB, int N) {
    int i = blockIdx.x * 256 + threadIdx.x;
    if (i >= N * D_FEAT) return;
    int n = i / D_FEAT, c = i - n * D_FEAT;
    float v = (c < D_IN) ? x[(size_t)n * D_IN + c] : deg[(size_t)n * D_DEG + (c - D_IN)];
    hB[i] = f2bf(v);
}

// W [dIn][dOut] f32  ->  WT [dOut][kPad] bf16, zero-padded for k >= dIn
__global__ void transpose_w_kernel(const float* __restrict__ W,
                                   unsigned short* __restrict__ WT,
                                   int dIn, int dOut, int kPad) {
    int i = blockIdx.x * 256 + threadIdx.x;
    if (i >= dOut * kPad) return;
    int o = i / kPad, k = i - o * kPad;
    WT[i] = (k < dIn) ? f2bf(W[(size_t)k * dOut + o]) : (unsigned short)0;
}

// ------------------ persistent edge message MLP + scatter ------------------

#define MSG_NT 512                       // 16 waves
#define CHUNKS (MT * (K_A / 8))          // 32 rows * 36 x 16B chunks = 1152

struct SmemMsg {
    unsigned short w1[D_HID * K_A];      // 147456 B : msg_W1^T  [256][288]
    unsigned short w2[D_MSG * D_HID];    //  65536 B : msg_W2^T  [128][256]
    unsigned short a[2][MT * K_A];       //  36864 B : double-buffered edge tile
    unsigned short h[MT * D_HID];        //  16384 B : relu hidden (bf16)
    int tgt[2][MT];
};                                       // ~266 KB of the 320 KB WGP LDS

__global__ __launch_bounds__(MSG_NT)
void msg_kernel(const unsigned short* __restrict__ hB,      // [N][144] bf16
                const long long* __restrict__ edge,         // [2][E] (src row0, tgt row1)
                const unsigned short* __restrict__ W1T,     // [256][288] bf16
                const float* __restrict__ b1,               // [256]
                const unsigned short* __restrict__ W2T,     // [128][256] bf16
                const float* __restrict__ b2,               // [128]
                float* __restrict__ agg,                    // [N][128] f32 (atomic accum)
                int E, int nTiles) {
    __shared__ SmemMsg sm;

    const int tid  = threadIdx.x;
    const int lane = tid & 31;
    const int wv   = tid >> 5;                  // 0..15
    const int nc   = lane & 15;
    const int moff = (lane >> 4) ? 8 : 0;
    // per-wave async ops per staged tile (CHUNKS spread over MSG_NT threads)
    const int myAsync = (wv < (CHUNKS % MSG_NT) / 32) ? (CHUNKS / MSG_NT + 1)
                                                      : (CHUNKS / MSG_NT);   // 3 or 2

    // loop-invariant per-wave constants
    const int   mtG2   = wv >> 3;               // GEMM2 m-tile
    const int   ntG2   = wv & 7;                // GEMM2 n-tile
    const float biasG1 = b1[wv * 16 + nc];      // GEMM1 bias (col fixed per wave)
    const float biasG2 = b2[ntG2 * 16 + nc];    // GEMM2 bias

    // each thread owns one edge row of the tile (chunk-major staging)
    const int er = tid & 31;

    // ---- prologue: kick off async gather of first tile ----
    int t0 = blockIdx.x;
    if (t0 < nTiles) {
        const int eg  = t0 * MT + er;
        const int egc = (eg < E) ? eg : (E - 1);
        const long long s  = edge[egc];
        const long long tt = edge[(long long)E + egc];
        if (tid < MT) sm.tgt[0][er] = (int)tt;
        const unsigned short* rowT = hB + (size_t)tt * D_FEAT;
        const unsigned short* rowS = hB + (size_t)s  * D_FEAT;
        for (int i = tid; i < CHUNKS; i += MSG_NT) {
            const int c = i >> 5;               // 0..35
            const void* src = (c < 18) ? (const void*)(rowT + c * 8)
                                       : (const void*)(rowS + (c - 18) * 8);
            async_g2l_b128(sm.a[0] + er * K_A + c * 8, src);
        }
    }

    // ---- stage weights into LDS (overlaps with async gather) ----
    {
        const uint4* g1 = (const uint4*)W1T;
        uint4*       s1 = (uint4*)sm.w1;
        for (int i = tid; i < (D_HID * K_A) / 8; i += MSG_NT) s1[i] = g1[i];
        const uint4* g2 = (const uint4*)W2T;
        uint4*       s2 = (uint4*)sm.w2;
        for (int i = tid; i < (D_MSG * D_HID) / 8; i += MSG_NT) s2[i] = g2[i];
    }

    int it = 0;
    for (int t = t0; t < nTiles; t += gridDim.x, ++it) {
        const int  cur     = it & 1;
        const int  tNext   = t + (int)gridDim.x;
        const bool hasNext = tNext < nTiles;

        // ---- issue async gather for the NEXT tile into the other buffer ----
        // (safe: sm.a[1-cur] was last read in GEMM1 of the previous iteration,
        //  which is sealed by that iteration's post-GEMM1 barrier; sm.tgt[1-cur]
        //  is only read from the register snapshot taken below, never live here)
        if (hasNext) {
            const int eg  = tNext * MT + er;
            const int egc = (eg < E) ? eg : (E - 1);
            const long long s  = edge[egc];
            const long long tt = edge[(long long)E + egc];
            if (tid < MT) sm.tgt[1 - cur][er] = (int)tt;
            const unsigned short* rowT = hB + (size_t)tt * D_FEAT;
            const unsigned short* rowS = hB + (size_t)s  * D_FEAT;
            for (int i = tid; i < CHUNKS; i += MSG_NT) {
                const int c = i >> 5;
                const void* src = (c < 18) ? (const void*)(rowT + c * 8)
                                           : (const void*)(rowS + (c - 18) * 8);
                async_g2l_b128(sm.a[1 - cur] + er * K_A + c * 8, src);
            }
        }

        // wait for CURRENT buffer's async ops (in-order completion), keep next in flight
        wait_asynccnt_le(hasNext ? myAsync : 0);
        __syncthreads();

        const unsigned short* lA = sm.a[cur];
        const int eBase = t * MT;

        // ---- snapshot this wave's scatter targets into registers ----
        // Reads of sm.tgt[cur] are race-free here: the next write to this buffer
        // (staging of iteration it+2... i.e. tgt[cur]) only happens after every
        // wave passed this iteration's post-GEMM1 barrier below.
        int tgtReg[8];
        #pragma unroll
        for (int r = 0; r < 8; ++r) tgtReg[r] = sm.tgt[cur][mtG2 * 16 + r + moff];

        // ---- GEMM1: [32 x 288] x [288 x 256] -> relu -> sm.h (bf16) ----
        {
            v8f acc[2] = {};                        // 2 m-tiles, 1 n-tile per wave
            const unsigned short* wb = sm.w1 + (size_t)(wv * 16) * K_A;
            #pragma unroll
            for (int ks = 0; ks < K_A / 32; ++ks) {
                Frag a0 = load_a(lA + ks * 32, K_A, lane);
                Frag a1 = load_a(lA + 16 * K_A + ks * 32, K_A, lane);
                Frag bb = load_b(wb + ks * 32, K_A, lane);
                acc[0] = wmma_bf16(a0, bb, acc[0]);
                acc[1] = wmma_bf16(a1, bb, acc[1]);
            }
            const int col = wv * 16 + nc;
            #pragma unroll
            for (int r = 0; r < 8; ++r) {
                #pragma unroll
                for (int mt = 0; mt < 2; ++mt) {
                    const int m = mt * 16 + r + moff;
                    float h0 = acc[mt][r] + biasG1;
                    h0 = h0 > 0.f ? h0 : 0.f;
                    sm.h[m * D_HID + col] = f2bf(h0);
                }
            }
        }
        __syncthreads();   // sm.h ready; all waves done reading sm.a[cur]/sm.tgt[cur]

        // ---- GEMM2: [32 x 256] x [256 x 128] -> +bias -> atomic scatter ----
        {
            v8f acc = {};
            const unsigned short* wb = sm.w2 + (size_t)(ntG2 * 16) * D_HID;
            const unsigned short* ha = sm.h + mtG2 * 16 * D_HID;
            #pragma unroll
            for (int ks = 0; ks < D_HID / 32; ++ks) {
                Frag a  = load_a(ha + ks * 32, D_HID, lane);
                Frag bb = load_b(wb + ks * 32, D_HID, lane);
                acc = wmma_bf16(a, bb, acc);
            }
            const int col = ntG2 * 16 + nc;
            #pragma unroll
            for (int r = 0; r < 8; ++r) {
                const int m  = mtG2 * 16 + r + moff;
                const int eg = eBase + m;
                if (eg < E) {
                    atomicAdd(agg + (size_t)tgtReg[r] * D_MSG + col, acc[r] + biasG2);
                }
            }
        }
        // no trailing barrier needed: next iteration's top-of-loop barrier
        // protects sm.h, and sm.a/sm.tgt staging targets the other buffer.
    }
}

// ------------------------------ node update MLP ----------------------------

__global__ __launch_bounds__(256)
void upd_kernel(const unsigned short* __restrict__ hB,      // [N][144] bf16
                const float* __restrict__ agg,              // [N][128] f32
                const unsigned short* __restrict__ U1T,     // [256][288] bf16 (K 272->288 pad)
                const float* __restrict__ b1,               // [256]
                const unsigned short* __restrict__ U2T,     // [128][256] bf16
                const float* __restrict__ b2,               // [128]
                float* __restrict__ out,                    // [N][128] f32
                int N) {
    __shared__ unsigned short lA[MT * K_A];
    __shared__ unsigned short lH[MT * D_HID];

    const int tid   = threadIdx.x;
    const int lane  = tid & 31;
    const int wv    = tid >> 5;
    const int nBase = blockIdx.x * MT;

    // ---- stage 32 nodes: [h | bf16(agg) | zero pad(16)] ----
    {
        const int nr = tid >> 3;
        const int l8 = tid & 7;
        const int ng = nBase + nr;
        unsigned short* dst = lA + nr * K_A;
        if (ng < N) {
            const uint4* rH = (const uint4*)(hB + (size_t)ng * D_FEAT);
            uint4* d4 = (uint4*)dst;
            for (int i = l8; i < 18; i += 8) d4[i] = rH[i];          // cols 0..143
            const float* ar = agg + (size_t)ng * D_MSG;
            unsigned int* d2 = (unsigned int*)(dst + D_FEAT);        // cols 144..271
            for (int i = l8; i < 64; i += 8) {
                unsigned lo = f2bf(ar[2 * i]);
                unsigned hi = f2bf(ar[2 * i + 1]);
                d2[i] = lo | (hi << 16);
            }
            for (int i = l8; i < 16; i += 8) dst[272 + i] = 0;       // cols 272..287 pad
        } else {
            const uint4 z = make_uint4(0u, 0u, 0u, 0u);
            uint4* d4 = (uint4*)dst;
            for (int i = l8; i < 36; i += 8) d4[i] = z;
        }
    }
    __syncthreads();

    // ---- GEMM1: [32 x 288] x [288 x 256] -> relu -> lH ----
    {
        v8f acc[2][2] = {};
        const int colGrp = wv * 32;
        #pragma unroll
        for (int ks = 0; ks < K_A / 32; ++ks) {
            Frag a0  = load_a(lA + ks * 32, K_A, lane);
            Frag a1  = load_a(lA + 16 * K_A + ks * 32, K_A, lane);
            Frag bb0 = load_b(U1T + (size_t)colGrp * K_A + ks * 32, K_A, lane);
            Frag bb1 = load_b(U1T + (size_t)(colGrp + 16) * K_A + ks * 32, K_A, lane);
            acc[0][0] = wmma_bf16(a0, bb0, acc[0][0]);
            acc[0][1] = wmma_bf16(a0, bb1, acc[0][1]);
            acc[1][0] = wmma_bf16(a1, bb0, acc[1][0]);
            acc[1][1] = wmma_bf16(a1, bb1, acc[1][1]);
        }
        const int   nc   = lane & 15;
        const int   moff = (lane >> 4) ? 8 : 0;
        const float bi0  = b1[colGrp + nc];
        const float bi1  = b1[colGrp + 16 + nc];
        #pragma unroll
        for (int r = 0; r < 8; ++r) {
            #pragma unroll
            for (int mt = 0; mt < 2; ++mt) {
                const int m = mt * 16 + r + moff;
                float h0 = acc[mt][0][r] + bi0; h0 = h0 > 0.f ? h0 : 0.f;
                float h1 = acc[mt][1][r] + bi1; h1 = h1 > 0.f ? h1 : 0.f;
                lH[m * D_HID + colGrp + nc]      = f2bf(h0);
                lH[m * D_HID + colGrp + 16 + nc] = f2bf(h1);
            }
        }
    }
    __syncthreads();

    // ---- GEMM2: [32 x 256] x [256 x 128] -> +bias -> out ----
    {
        v8f acc[2] = {};
        const int colG = wv * 16;
        #pragma unroll
        for (int ks = 0; ks < D_HID / 32; ++ks) {
            Frag a0 = load_a(lH + ks * 32, D_HID, lane);
            Frag a1 = load_a(lH + 16 * D_HID + ks * 32, D_HID, lane);
            Frag bb = load_b(U2T + (size_t)colG * D_HID + ks * 32, D_HID, lane);
            acc[0] = wmma_bf16(a0, bb, acc[0]);
            acc[1] = wmma_bf16(a1, bb, acc[1]);
        }
        const int   nc   = lane & 15;
        const int   moff = (lane >> 4) ? 8 : 0;
        const float bia  = b2[colG + nc];
        #pragma unroll
        for (int mt = 0; mt < 2; ++mt) {
            #pragma unroll
            for (int r = 0; r < 8; ++r) {
                const int m  = mt * 16 + r + moff;
                const int ng = nBase + m;
                if (ng < N) {
                    out[(size_t)ng * D_UP + colG + nc] = acc[mt][r] + bia;
                }
            }
        }
    }
}

// --------------------------------- launcher --------------------------------

extern "C" void kernel_launch(void* const* d_in, const int* in_sizes, int n_in,
                              void* d_out, int out_size, void* d_ws, size_t ws_size,
                              hipStream_t stream) {
    (void)n_in; (void)out_size; (void)ws_size;
    const float*     x   = (const float*)d_in[0];
    const long long* ei  = (const long long*)d_in[1];
    const float*     deg = (const float*)d_in[2];
    const float*     mW1 = (const float*)d_in[3];
    const float*     mb1 = (const float*)d_in[4];
    const float*     mW2 = (const float*)d_in[5];
    const float*     mb2 = (const float*)d_in[6];
    const float*     uW1 = (const float*)d_in[7];
    const float*     ub1 = (const float*)d_in[8];
    const float*     uW2 = (const float*)d_in[9];
    const float*     ub2 = (const float*)d_in[10];

    const int N = in_sizes[0] / D_IN;
    const int E = in_sizes[1] / 2;

    // workspace carve-out (256 B aligned)
    char*  ws  = (char*)d_ws;
    size_t off = 0;
    auto carve = [&](size_t bytes) {
        char* p = ws + off;
        off = (off + bytes + 255) & ~(size_t)255;
        return p;
    };
    unsigned short* hB  = (unsigned short*)carve((size_t)N * D_FEAT * 2);
    float*          agg = (float*)carve((size_t)N * D_MSG * 4);
    unsigned short* W1T = (unsigned short*)carve((size_t)D_HID * K_A * 2);
    unsigned short* W2T = (unsigned short*)carve((size_t)D_MSG * D_HID * 2);
    unsigned short* U1T = (unsigned short*)carve((size_t)D_HID * K_A * 2);
    unsigned short* U2T = (unsigned short*)carve((size_t)D_UP * D_HID * 2);

    hipMemsetAsync(agg, 0, (size_t)N * D_MSG * 4, stream);

    build_h_kernel<<<(N * D_FEAT + 255) / 256, 256, 0, stream>>>(x, deg, hB, N);
    transpose_w_kernel<<<(D_HID * K_A + 255) / 256, 256, 0, stream>>>(mW1, W1T, K_A, D_HID, K_A);
    transpose_w_kernel<<<(D_MSG * D_HID + 255) / 256, 256, 0, stream>>>(mW2, W2T, D_HID, D_MSG, D_HID);
    transpose_w_kernel<<<(D_HID * K_A + 255) / 256, 256, 0, stream>>>(uW1, U1T, 272, D_HID, K_A);
    transpose_w_kernel<<<(D_UP * D_HID + 255) / 256, 256, 0, stream>>>(uW2, U2T, D_HID, D_UP, D_HID);

    const int nTiles  = (E + MT - 1) / MT;
    const int msgGrid = (nTiles < 1024) ? nTiles : 1024;
    msg_kernel<<<msgGrid, MSG_NT, 0, stream>>>(hB, ei, W1T, mb1, W2T, mb2, agg, E, nTiles);

    upd_kernel<<<(N + MT - 1) / MT, 256, 0, stream>>>(hB, agg, U1T, ub1, U2T, ub2,
                                                      (float*)d_out, N);
}
